// XdiffMultiHeadAttention_79791902425494
// MI455X (gfx1250) — matrix operand order repaired and checked by reading, not compiled, not run on hardware
//
#include <hip/hip_runtime.h>
#include <hip/hip_bf16.h>

// ---------------------------------------------------------------------------
// Types / helpers
// ---------------------------------------------------------------------------
typedef __attribute__((ext_vector_type(16))) __bf16 v16bf;
typedef __attribute__((ext_vector_type(8)))  float  v8f;

#define S_LEN 2048
#define D_MOD 512
#define NHEAD 8
#define DHEAD 64
#define BATCH 2
#define PI_F  3.14159265358979323846f

__device__ __forceinline__ __bf16 f2bf(float f) {
    unsigned u = __builtin_bit_cast(unsigned, f);
    unsigned r = u + 0x7FFFu + ((u >> 16) & 1u);   // round-to-nearest-even
    unsigned short h = (unsigned short)(r >> 16);
    return __builtin_bit_cast(__bf16, h);
}
__device__ __forceinline__ float bf2f(__bf16 b) {
    unsigned short h = __builtin_bit_cast(unsigned short, b);
    unsigned u = ((unsigned)h) << 16;
    return __builtin_bit_cast(float, u);
}

// 10-basis bias:  c0*x + c1*(-x^2/2) + sum c_i * sin/cos(n*pi*x), n in {1,3,5,9}
// harmonics via angle-addition chain from sin/cos(pi x) (pure FMA, no extra trans)
__device__ __forceinline__ float basis10(float x, const float* cf) {
    float s1, c1;
    __sincosf(PI_F * x, &s1, &c1);
    const float s2 = 2.0f * s1 * c1;
    const float c2 = c1 * c1 - s1 * s1;
    const float s3 = s1 * c2 + c1 * s2, c3 = c1 * c2 - s1 * s2;
    const float s5 = s3 * c2 + c3 * s2, c5 = c3 * c2 - s3 * s2;
    const float s7 = s5 * c2 + c5 * s2, c7 = c5 * c2 - s5 * s2;
    const float s9 = s7 * c2 + c7 * s2, c9 = c7 * c2 - s7 * s2;
    return cf[0] * x + cf[1] * (-0.5f * x * x)
         + cf[2] * s1 + cf[3] * c1 + cf[4] * s3 + cf[5] * c3
         + cf[6] * s5 + cf[7] * c5 + cf[8] * s9 + cf[9] * c9;
}

// ---------------------------------------------------------------------------
// Generic tiled GEMM:  Out(MxN) = A(MxK) * W(KxN) + bias
// A fp32 (converted) or bf16; W/bias fp32; Out fp32 or bf16.
// 256 threads (8 waves), 64x64 tile, 2 persistent accumulators per wave.
// W tile staged in LDS transposed (pitch 40 -> 16B-aligned rows, conflict-free:
// bank base (c*20)%64 distinct for 16 lanes) so B frags are 2x ds_load_b128.
// Fragment layouts per CDNA5 ISA 7.12.2 (wave32):
//   A elem i : K = ((i&8)? i+8 : i) + 8*half ;  M = lane%16
//   B elem i : K = ((i&8)? i+8 : i) + 8*half              ;  N = lane%16
//   D elem r : M = r + 8*half               ;  N = lane%16
// ---------------------------------------------------------------------------
#define WPITCH 40
template <bool ABF16, bool OUTBF16>
__global__ __launch_bounds__(256) void gemm_wmma_kernel(
        const void* __restrict__ Ap, const float* __restrict__ Wp,
        const float* __restrict__ bias, void* __restrict__ Outp,
        int M, int N, int K) {
    __shared__ __bf16 wt[64 * WPITCH];   // [col][krow], pitch 40
    const float*  Af = (const float*)Ap;
    const __bf16* Ab = (const __bf16*)Ap;
    const int tid  = threadIdx.x;
    const int wave = tid >> 5;
    const int lane = tid & 31;
    const int half = lane >> 4;
    const int lm   = lane & 15;
    const int row0 = blockIdx.x * 64;
    const int col0 = blockIdx.y * 64;

    const int r0a = row0 + ((wave >> 2) + 0) * 16, c0a = (wave & 3) * 16;
    const int r0b = row0 + ((wave >> 2) + 2) * 16, c0b = (wave & 3) * 16;
    v8f acc0 = {}, acc1 = {};

    for (int kc = 0; kc < K; kc += 32) {
        // ---- stage W tile (32 x 64) coalesced, store transposed -------------
        for (int idx = tid; idx < 32 * 64; idx += 256) {
            const int r = idx >> 6, c = idx & 63;
            wt[c * WPITCH + r] = f2bf(Wp[(size_t)(kc + r) * N + col0 + c]);
        }
        __syncthreads();

        v16bf aa, ab, b0, b1;
#pragma unroll
        for (int i = 0; i < 16; ++i) {
            const int ka = ((i & 8) ? (i + 8) : i) + half * 8 + kc;
            if (ABF16) {
                aa[i] = Ab[(size_t)(r0a + lm) * K + ka];
                ab[i] = Ab[(size_t)(r0b + lm) * K + ka];
            } else {
                aa[i] = f2bf(Af[(size_t)(r0a + lm) * K + ka]);
                ab[i] = f2bf(Af[(size_t)(r0b + lm) * K + ka]);
            }
            b0[i] = wt[(c0a + lm) * WPITCH + half * 8 + ((i & 8) ? (i + 8) : i)];
            b1[i] = wt[(c0b + lm) * WPITCH + half * 8 + ((i & 8) ? (i + 8) : i)];
        }
        acc0 = __builtin_amdgcn_wmma_f32_16x16x32_bf16(
            false, aa, false, b0, (short)0, acc0, false, false);
        acc1 = __builtin_amdgcn_wmma_f32_16x16x32_bf16(
            false, ab, false, b1, (short)0, acc1, false, false);
        __syncthreads();
    }
#pragma unroll
    for (int r = 0; r < 8; ++r) {
        const int rowA = r0a + r + half * 8, colA = col0 + c0a + lm;
        const int rowB = r0b + r + half * 8, colB = col0 + c0b + lm;
        const float vA = acc0[r] + bias[colA];
        const float vB = acc1[r] + bias[colB];
        if (OUTBF16) {
            ((__bf16*)Outp)[(size_t)rowA * N + colA] = f2bf(vA);
            ((__bf16*)Outp)[(size_t)rowB * N + colB] = f2bf(vB);
        } else {
            ((float*)Outp)[(size_t)rowA * N + colA] = vA;
            ((float*)Outp)[(size_t)rowB * N + colB] = vB;
        }
    }
}

// ---------------------------------------------------------------------------
// Coefficient kernel: coefs[n,b,h,s] = scale * (qp[b,s,:] . Wcoef[n,:,h] + bcoef[n,h])
// scale folds the 0.1 einsum factor and 1/sqrt(dh); abs() on n==1.
// ---------------------------------------------------------------------------
__global__ __launch_bounds__(256) void coef_kernel(
        const __bf16* __restrict__ qp, const float* __restrict__ Wc,
        const float* __restrict__ bc, float* __restrict__ coefs) {
    const int idx = blockIdx.x * 256 + threadIdx.x;  // (((n*B+b)*H+h)*S + s)
    const int s = idx & (S_LEN - 1);
    int t = idx >> 11;
    const int h = t & (NHEAD - 1); t >>= 3;
    const int b = t & 1;
    const int n = t >> 1;

    const __bf16* qrow = qp + ((size_t)(b * S_LEN + s)) * D_MOD;
    const float*  w    = Wc + (size_t)n * D_MOD * NHEAD + h;
    float acc = bc[n * NHEAD + h];
    for (int d = 0; d < D_MOD; ++d)
        acc += bf2f(qrow[d]) * w[(size_t)d * NHEAD];
    if (n == 1) acc = fabsf(acc);
    coefs[idx] = acc * 0.0125f;   // 0.1 * (1/8)
}

// ---------------------------------------------------------------------------
// Fused attention kernel. One WG = (b, h, 16 query rows). 256 threads, 8 waves.
// LDS: 16x2048 fp32 logit strip (131 KB of the WGP's 320 KB -> 2 WGs/WGP).
//  P1: QK^T, two k-tiles in flight per wave (fills WMMA hazard slots)
//  P2: harmonic-basis bias + softmax, float4-vectorized streams
//  P3: coalesced float4 normalized attn write
//  P4: P.V via bf16 WMMA, cross-wave ds_add_f32 reduce, bf16 ctx to ws
// ---------------------------------------------------------------------------
__global__ __launch_bounds__(256) void attn_kernel(
        const __bf16* __restrict__ qp, const __bf16* __restrict__ kp,
        const __bf16* __restrict__ vp, const float* __restrict__ xdiff,
        const float* __restrict__ coefs, float* __restrict__ attn_out,
        __bf16* __restrict__ ctx) {
    __shared__ float lg[16 * S_LEN];
    __shared__ float ctx_s[16 * DHEAD];
    __shared__ float inv_s[16];

    const int h  = blockIdx.x;           // fastest: keeps xdiff L2-hot across heads
    const int q0 = blockIdx.y * 16;
    const int b  = blockIdx.z;
    const int tid  = threadIdx.x;
    const int wave = tid >> 5;
    const int lane = tid & 31;
    const int half = lane >> 4;
    const int lm   = lane & 15;

    // ---- Phase 1: logits = (Q K^T) / 8 -> LDS -------------------------------
    {
        v16bf aq0, aq1;                  // A frags for both dh chunks (row fixed)
#pragma unroll
        for (int i = 0; i < 16; ++i) {
            const int ka = ((i & 8) ? (i + 8) : i) + half * 8;
            const size_t base = (size_t)(b * S_LEN + q0 + lm) * D_MOD + h * DHEAD;
            aq0[i] = qp[base + ka];
            aq1[i] = qp[base + 32 + ka];
        }
        for (int kt = wave * 2; kt < S_LEN / 16; kt += 16) {
            v8f A0 = {}, A1 = {};
#pragma unroll
            for (int chunk = 0; chunk < 2; ++chunk) {
                v16bf b0, b1;
#pragma unroll
                for (int i = 0; i < 16; ++i) {
                    const int kk = chunk * 32 + half * 8 + ((i & 8) ? (i + 8) : i);
                    b0[i] = kp[(size_t)(b * S_LEN + kt * 16 + lm) * D_MOD
                               + h * DHEAD + kk];
                    b1[i] = kp[(size_t)(b * S_LEN + (kt + 1) * 16 + lm) * D_MOD
                               + h * DHEAD + kk];
                }
                const v16bf a = chunk ? aq1 : aq0;
                A0 = __builtin_amdgcn_wmma_f32_16x16x32_bf16(
                    false, a, false, b0, (short)0, A0, false, false);
                A1 = __builtin_amdgcn_wmma_f32_16x16x32_bf16(
                    false, a, false, b1, (short)0, A1, false, false);
            }
#pragma unroll
            for (int r = 0; r < 8; ++r) {
                lg[(r + half * 8) * S_LEN + kt * 16 + lm]       = A0[r] * 0.125f;
                lg[(r + half * 8) * S_LEN + (kt + 1) * 16 + lm] = A1[r] * 0.125f;
            }
        }
    }
    for (int i = tid; i < 16 * DHEAD; i += 256) ctx_s[i] = 0.0f;
    __syncthreads();

    // ---- Phase 2: add basis bias, softmax (16 lanes cooperate per row) ------
    {
        const int q = tid >> 4, sub = tid & 15;
        float cf[10];
#pragma unroll
        for (int n = 0; n < 10; ++n)
            cf[n] = coefs[(size_t)(((n * BATCH + b) * NHEAD + h)) * S_LEN + q0 + q];
        const float* xrow = xdiff + ((size_t)(b * S_LEN + q0 + q)) * S_LEN;
        float* row = lg + q * S_LEN;
        const int k0 = sub * 128;

        float mx = -1.0e30f;
        for (int k = k0; k < k0 + 128; k += 4) {
            const float4 x4 = *(const float4*)(xrow + k);
            float4 l4 = *(float4*)(row + k);
            l4.x += basis10(x4.x, cf);
            l4.y += basis10(x4.y, cf);
            l4.z += basis10(x4.z, cf);
            l4.w += basis10(x4.w, cf);
            *(float4*)(row + k) = l4;
            mx = fmaxf(mx, fmaxf(fmaxf(l4.x, l4.y), fmaxf(l4.z, l4.w)));
        }
#pragma unroll
        for (int m = 8; m >= 1; m >>= 1) mx = fmaxf(mx, __shfl_xor(mx, m, 16));

        float sum = 0.0f;
        for (int k = k0; k < k0 + 128; k += 4) {
            float4 l4 = *(float4*)(row + k);
            l4.x = __expf(l4.x - mx);
            l4.y = __expf(l4.y - mx);
            l4.z = __expf(l4.z - mx);
            l4.w = __expf(l4.w - mx);
            *(float4*)(row + k) = l4;
            sum += (l4.x + l4.y) + (l4.z + l4.w);
        }
#pragma unroll
        for (int m = 8; m >= 1; m >>= 1) sum += __shfl_xor(sum, m, 16);
        if (sub == 0) inv_s[q] = 1.0f / sum;
    }
    __syncthreads();

    // ---- Phase 3: coalesced normalized attn write (float4) ------------------
    for (int q = 0; q < 16; ++q) {
        const float inv = inv_s[q];
        float4* dst = (float4*)(attn_out
                      + ((size_t)((b * NHEAD + h) * S_LEN + q0 + q)) * S_LEN);
        const float4* src = (const float4*)(lg + q * S_LEN);
        for (int k4 = tid; k4 < S_LEN / 4; k4 += 256) {
            float4 v = src[k4];
            v.x *= inv; v.y *= inv; v.z *= inv; v.w *= inv;
            dst[k4] = v;
        }
    }

    // ---- Phase 4: ctx = P . V (bf16 WMMA, cross-wave LDS reduce) ------------
    {
        v8f zero = {};
        v8f acc[4] = {zero, zero, zero, zero};
        for (int cc = 0; cc < 8; ++cc) {
            const int chunk = wave * 8 + cc;             // 64 key-chunks of 32
            v16bf a;
#pragma unroll
            for (int i = 0; i < 16; ++i) {
                const int kk = ((i & 8) ? (i + 8) : i) + half * 8 + chunk * 32;
                a[i] = f2bf(lg[lm * S_LEN + kk]);        // row lm = query row
            }
#pragma unroll
            for (int nt = 0; nt < 4; ++nt) {
                v16bf bb;
#pragma unroll
                for (int i = 0; i < 16; ++i)
                    bb[i] = vp[(size_t)(b * S_LEN + chunk * 32 + half * 8 + ((i & 8) ? (i + 8) : i)) * D_MOD
                               + h * DHEAD + nt * 16 + lm];
                acc[nt] = __builtin_amdgcn_wmma_f32_16x16x32_bf16(
                    false, a, false, bb, (short)0, acc[nt], false, false);
            }
        }
#pragma unroll
        for (int nt = 0; nt < 4; ++nt)
#pragma unroll
            for (int r = 0; r < 8; ++r)
                atomicAdd(&ctx_s[(r + half * 8) * DHEAD + nt * 16 + lm], acc[nt][r]);
    }
    __syncthreads();

    // ---- Phase 5: normalized context -> bf16 workspace ----------------------
    for (int i = tid; i < 16 * DHEAD; i += 256) {
        const int q = i >> 6, d = i & 63;
        ctx[(size_t)(b * S_LEN + q0 + q) * D_MOD + h * DHEAD + d] =
            f2bf(ctx_s[i] * inv_s[q]);
    }
}

// ---------------------------------------------------------------------------
// Host launcher
// ---------------------------------------------------------------------------
extern "C" void kernel_launch(void* const* d_in, const int* in_sizes, int n_in,
                              void* d_out, int out_size, void* d_ws, size_t ws_size,
                              hipStream_t stream) {
    const float* q     = (const float*)d_in[0];
    const float* k     = (const float*)d_in[1];
    const float* v     = (const float*)d_in[2];
    const float* xdiff = (const float*)d_in[3];
    const float* Wq    = (const float*)d_in[4];
    const float* bq    = (const float*)d_in[5];
    const float* Wk    = (const float*)d_in[6];
    const float* bk    = (const float*)d_in[7];
    const float* Wv    = (const float*)d_in[8];
    const float* bv    = (const float*)d_in[9];
    const float* Wcoef = (const float*)d_in[10];
    const float* bcoef = (const float*)d_in[11];
    const float* Wo    = (const float*)d_in[12];
    const float* bo    = (const float*)d_in[13];

    const size_t M = (size_t)BATCH * S_LEN;             // 4096
    char* ws = (char*)d_ws;
    const size_t projB = M * D_MOD * sizeof(__bf16);    // 4 MiB each
    __bf16* qp  = (__bf16*)(ws);
    __bf16* kp  = (__bf16*)(ws + projB);
    __bf16* vp  = (__bf16*)(ws + 2 * projB);
    __bf16* ctx = (__bf16*)(ws + 3 * projB);
    float*  coefs = (float*)(ws + 4 * projB);           // 10*B*H*S fp32

    dim3 gg(M / 64, D_MOD / 64);                        // 64 x 8
    gemm_wmma_kernel<false, true><<<gg, 256, 0, stream>>>(q, Wq, bq, qp,
                                                          (int)M, D_MOD, D_MOD);
    gemm_wmma_kernel<false, true><<<gg, 256, 0, stream>>>(k, Wk, bk, kp,
                                                          (int)M, D_MOD, D_MOD);
    gemm_wmma_kernel<false, true><<<gg, 256, 0, stream>>>(v, Wv, bv, vp,
                                                          (int)M, D_MOD, D_MOD);

    const int nCoef = 10 * BATCH * NHEAD * S_LEN;       // 327680
    coef_kernel<<<nCoef / 256, 256, 0, stream>>>(qp, Wcoef, bcoef, coefs);

    float* out_p  = (float*)d_out;                      // (B,S,D)
    float* attn_p = out_p + M * D_MOD;                  // (B,H,S,S)
    dim3 ga(NHEAD, S_LEN / 16, BATCH);                  // h fastest -> xdiff L2 reuse
    attn_kernel<<<ga, 256, 0, stream>>>(qp, kp, vp, xdiff, coefs, attn_p, ctx);

    gemm_wmma_kernel<true, false><<<gg, 256, 0, stream>>>(ctx, Wo, bo, out_p,
                                                          (int)M, D_MOD, D_MOD);
}
